// SelfAttentionDecoder_54924041781444
// MI455X (gfx1250) — matrix operand order repaired and checked by reading, not compiled, not run on hardware
//
#include <hip/hip_runtime.h>

typedef __attribute__((ext_vector_type(16))) _Float16 v16h;
typedef __attribute__((ext_vector_type(8)))  _Float16 v8h;
typedef __attribute__((ext_vector_type(8)))  float    v8f;
typedef __attribute__((ext_vector_type(4)))  unsigned int v4u;
typedef __attribute__((ext_vector_type(8)))  int      v8i;
typedef __attribute__((ext_vector_type(4)))  int      v4i;

#define B_  4
#define S_  2048
#define D_  768
#define H_  12
#define HD_ 64
#define M_  (B_*S_)    // 8192 rows
#define N3_ (3*D_)     // 2304

// ---------------------------------------------------------------------------
// WMMA helper: D = A(16x32 f16) * B(32x16 f16) + C(16x16 f32)
// ---------------------------------------------------------------------------
__device__ __forceinline__ v8f wmma_f16(v16h a, v16h b, v8f c) {
  return __builtin_amdgcn_wmma_f32_16x16x32_f16(
      false, a, false, b, (short)0, c, false, false);
}

// A-operand loader (contiguous k order, matching B):
//   elements 0..15 = K (half*16 + 0..15)
__device__ __forceinline__ v16h load_a16(const _Float16* __restrict__ row, int half) {
  v16h a;
  *(v8h*)&a       = *(const v8h*)(row + half * 16);
  *((v8h*)&a + 1) = *(const v8h*)(row + half * 16 + 8);
  return a;
}

// Flat pointer -> 32-bit LDS offset (flat LDS addr low 32 bits == DS offset)
__device__ __forceinline__ unsigned lds_addr32(const void* p) {
  return (unsigned)(uintptr_t)p;
}

// ---------------------------------------------------------------------------
// Tensor Data Mover: 2-D tile load, descriptor per ISA ch.8 (D# groups 0/1).
//   data_size = 2 bytes; rows (dim1) packed back-to-back into LDS.
//   amdgpu-toolchain (clang-23) 6-arg builtin form.
// ---------------------------------------------------------------------------
__device__ __forceinline__ void tdm_load_2d(unsigned lds_off, const void* gptr,
                                            unsigned tile_d0, unsigned tile_d1,
                                            unsigned tensor_d0, unsigned tensor_d1,
                                            unsigned stride0_elems) {
  const unsigned long long ga = (unsigned long long)(uintptr_t)gptr;
  v4u g0;
  g0.x = 1u;                                   // count=1, user descriptor
  g0.y = lds_off;                              // lds_addr (bytes)
  g0.z = (unsigned)ga;                         // global_addr[31:0]
  g0.w = (unsigned)(ga >> 32) | (2u << 30);    // global_addr[56:32] | type=2
  v8i g1;
  g1[0] = (int)(1u << 16);                     // workgroup_mask=0, data_size=1 (2B)
  g1[1] = (int)((tensor_d0 & 0xFFFFu) << 16);  // tensor_dim0[15:0]
  g1[2] = (int)((tensor_d0 >> 16) | ((tensor_d1 & 0xFFFFu) << 16));
  g1[3] = (int)((tensor_d1 >> 16) | (tile_d0 << 16));   // tile_dim0
  g1[4] = (int)(tile_d1 & 0xFFFFu);            // tile_dim1 (tile_dim2=0)
  g1[5] = (int)stride0_elems;                  // tensor_dim0_stride[31:0]
  g1[6] = 0;                                   // stride hi, dim1_stride lo
  g1[7] = 0;
  const v4i z4 = {0, 0, 0, 0};
  const v8i z8 = {0, 0, 0, 0, 0, 0, 0, 0};
  __builtin_amdgcn_tensor_load_to_lds(g0, g1, z4, z4, z8, 0);
}

// ---------------------------------------------------------------------------
// Precision conversion kernels
// ---------------------------------------------------------------------------
__global__ void cvt_f32_f16(const float* __restrict__ src, _Float16* __restrict__ dst, int n) {
  int i = blockIdx.x * blockDim.x + threadIdx.x;
  if (i < n) dst[i] = (_Float16)src[i];
}

// src[K][N] f32 -> dst[N][K] f16
__global__ void transpose_cvt(const float* __restrict__ src, _Float16* __restrict__ dst,
                              int K, int N) {
  int i = blockIdx.x * blockDim.x + threadIdx.x;
  if (i < K * N) {
    int n = i / K, k = i % K;
    dst[i] = (_Float16)src[(size_t)k * N + n];
  }
}

// ---------------------------------------------------------------------------
// GEMM core: one wave computes 16(M) x 64(N), K-loop of 32, explicit
// ping-pong unroll-by-2 (register sets A/B refilled directly by loads;
// no register-rotation moves).
// ---------------------------------------------------------------------------
__device__ __forceinline__ void gemm_16x64(const _Float16* __restrict__ arow,
                                           const _Float16* __restrict__ bbase,
                                           int col, int half, v8f acc[4]) {
  v16h aA = load_a16(arow, half);
  v16h bA0 = *(const v16h*)(bbase + (size_t)0 * 16 * D_);
  v16h bA1 = *(const v16h*)(bbase + (size_t)1 * 16 * D_);
  v16h bA2 = *(const v16h*)(bbase + (size_t)2 * 16 * D_);
  v16h bA3 = *(const v16h*)(bbase + (size_t)3 * 16 * D_);
  v16h aB, bB0, bB1, bB2, bB3;

  for (int kk = 0; kk < D_ - 64; kk += 64) {
    const int k1 = kk + 32, k2 = kk + 64;
    aB  = load_a16(arow + k1, half);
    bB0 = *(const v16h*)(bbase + (size_t)0 * 16 * D_ + k1);
    bB1 = *(const v16h*)(bbase + (size_t)1 * 16 * D_ + k1);
    bB2 = *(const v16h*)(bbase + (size_t)2 * 16 * D_ + k1);
    bB3 = *(const v16h*)(bbase + (size_t)3 * 16 * D_ + k1);
    acc[0] = wmma_f16(aA, bA0, acc[0]);
    acc[1] = wmma_f16(aA, bA1, acc[1]);
    acc[2] = wmma_f16(aA, bA2, acc[2]);
    acc[3] = wmma_f16(aA, bA3, acc[3]);
    aA  = load_a16(arow + k2, half);
    bA0 = *(const v16h*)(bbase + (size_t)0 * 16 * D_ + k2);
    bA1 = *(const v16h*)(bbase + (size_t)1 * 16 * D_ + k2);
    bA2 = *(const v16h*)(bbase + (size_t)2 * 16 * D_ + k2);
    bA3 = *(const v16h*)(bbase + (size_t)3 * 16 * D_ + k2);
    acc[0] = wmma_f16(aB, bB0, acc[0]);
    acc[1] = wmma_f16(aB, bB1, acc[1]);
    acc[2] = wmma_f16(aB, bB2, acc[2]);
    acc[3] = wmma_f16(aB, bB3, acc[3]);
  }
  {  // last pair: k-steps D_-64 (in A regs) and D_-32 (loaded into B regs)
    const int k1 = D_ - 32;
    aB  = load_a16(arow + k1, half);
    bB0 = *(const v16h*)(bbase + (size_t)0 * 16 * D_ + k1);
    bB1 = *(const v16h*)(bbase + (size_t)1 * 16 * D_ + k1);
    bB2 = *(const v16h*)(bbase + (size_t)2 * 16 * D_ + k1);
    bB3 = *(const v16h*)(bbase + (size_t)3 * 16 * D_ + k1);
    acc[0] = wmma_f16(aA, bA0, acc[0]);
    acc[1] = wmma_f16(aA, bA1, acc[1]);
    acc[2] = wmma_f16(aA, bA2, acc[2]);
    acc[3] = wmma_f16(aA, bA3, acc[3]);
    acc[0] = wmma_f16(aB, bB0, acc[0]);
    acc[1] = wmma_f16(aB, bB1, acc[1]);
    acc[2] = wmma_f16(aB, bB2, acc[2]);
    acc[3] = wmma_f16(aB, bB3, acc[3]);
  }
}

// ---------------------------------------------------------------------------
// QKV projection: X[8192,768]f16 @ WqkvT[2304,768]f16 + bias -> scatter
//   Q,K as [B,H,S,64] f16 ; V transposed to [B,H,64,S] f16
// ---------------------------------------------------------------------------
__global__ __launch_bounds__(256)
void qkv_gemm(const _Float16* __restrict__ X, const _Float16* __restrict__ Wt,
              const float* __restrict__ bias,
              _Float16* __restrict__ Qb, _Float16* __restrict__ Kb,
              _Float16* __restrict__ Vt) {
  const int lane = threadIdx.x & 31;
  const int wave = blockIdx.x * (blockDim.x >> 5) + (threadIdx.x >> 5);
  const int nT = N3_ / 64;
  const int mBase = (wave / nT) * 16;
  const int nBase = (wave % nT) * 64;
  const int col = lane & 15, half = lane >> 4;

  v8f acc[4];
#pragma unroll
  for (int t = 0; t < 4; ++t)
#pragma unroll
    for (int r = 0; r < 8; ++r) acc[t][r] = 0.0f;

  const _Float16* arow  = X + (size_t)(mBase + col) * D_;
  const _Float16* bbase = Wt + (size_t)(nBase + col) * D_ + half * 16;
  gemm_16x64(arow, bbase, col, half, acc);

#pragma unroll
  for (int t = 0; t < 4; ++t) {
    const int n = nBase + t * 16 + col;
    const float bv = bias[n];
    const int c3 = n / D_, rem = n % D_, h = rem / HD_, hd = rem % HD_;
#pragma unroll
    for (int r = 0; r < 8; ++r) {
      const int m = mBase + r + half * 8;
      const int b = m / S_, s = m % S_;
      const _Float16 hv = (_Float16)(acc[t][r] + bv);
      if (c3 == 0)      Qb[(((size_t)(b * H_ + h)) * S_ + s) * HD_ + hd] = hv;
      else if (c3 == 1) Kb[(((size_t)(b * H_ + h)) * S_ + s) * HD_ + hd] = hv;
      else              Vt[(((size_t)(b * H_ + h)) * HD_ + hd) * S_ + s] = hv;
    }
  }
}

// ---------------------------------------------------------------------------
// Flash-attention: block = 4 waves = 64 queries of one (b,h).
// K/V chunks (32 keys) DMA'd into LDS by the Tensor Data Mover, double-
// buffered (issue chunk c+1, s_wait_tensorcnt for chunk c, barrier, compute).
// ---------------------------------------------------------------------------
__global__ __launch_bounds__(128)
void attn(const _Float16* __restrict__ Qb, const _Float16* __restrict__ Kb,
          const _Float16* __restrict__ Vt, _Float16* __restrict__ Ao) {
  __shared__ __align__(128) _Float16 Klds[2][32 * HD_];   // [key][hd]
  __shared__ __align__(128) _Float16 Vlds[2][HD_ * 32];   // [hd][key]
  __shared__ __align__(32)  _Float16 pst[4][16 * 32];     // per-wave P staging

  const int tid = threadIdx.x;
  const int wid = tid >> 5;
  const int lane = tid & 31;
  const int col = lane & 15, half = lane >> 4;

  const int idx = blockIdx.x;
  const int qblk = idx & 31;            // S/64 = 32 query blocks
  const int bh = idx >> 5;
  const int h = bh % H_, b = bh / H_;
  const int qBase = qblk * 64 + wid * 16;
  const float scale = 0.125f;           // 64^-0.5

  const _Float16* Kbh = Kb + (size_t)bh * S_ * HD_;
  const _Float16* Vbh = Vt + (size_t)bh * HD_ * S_;

  const _Float16* qrow = Qb + ((size_t)bh * S_ + qBase + col) * HD_;
  const v16h aq0 = load_a16(qrow, half);        // hd 0..31
  const v16h aq1 = load_a16(qrow + 32, half);   // hd 32..63

  float m_run[8], l_run[8];
  v8f o[4];
#pragma unroll
  for (int r = 0; r < 8; ++r) { m_run[r] = -3.0e38f; l_run[r] = 0.0f; }
#pragma unroll
  for (int t = 0; t < 4; ++t)
#pragma unroll
    for (int r = 0; r < 8; ++r) o[t][r] = 0.0f;

  const int nCh = 2 * qblk + 2;   // causal: keys 0 .. qblk*64+63

  if (wid == 0) {  // prologue: DMA chunk 0 into buffer 0
    tdm_load_2d(lds_addr32(&Klds[0][0]), Kbh, HD_, 32, HD_, 32, HD_);
    tdm_load_2d(lds_addr32(&Vlds[0][0]), Vbh, 32, HD_, 32, HD_, S_);
  }

  for (int c = 0; c < nCh; ++c) {
    const int kb = c * 32;
    const int cur = c & 1;

    __syncthreads();   // all waves done reading buffer (cur^1) from iter c-1
    if (wid == 0) {
      if (c + 1 < nCh) {
        const int nb = cur ^ 1, kb2 = kb + 32;
        tdm_load_2d(lds_addr32(&Klds[nb][0]), Kbh + (size_t)kb2 * HD_, HD_, 32, HD_, 32, HD_);
        tdm_load_2d(lds_addr32(&Vlds[nb][0]), Vbh + kb2, 32, HD_, 32, HD_, S_);
        __builtin_amdgcn_s_wait_tensorcnt(2);   // chunk c (first 2 ops) complete
      } else {
        __builtin_amdgcn_s_wait_tensorcnt(0);
      }
    }
    __syncthreads();   // chunk c visible to all waves
    asm volatile("" ::: "memory");

    // ---- scores: 16 queries x 32 keys (contraction hd=64, 2 WMMAs/tile)
    v8f s0, s1;
#pragma unroll
    for (int r = 0; r < 8; ++r) { s0[r] = 0.0f; s1[r] = 0.0f; }
    const _Float16* kp0 = &Klds[cur][(size_t)col * HD_ + half * 16];
    const _Float16* kp1 = &Klds[cur][(size_t)(16 + col) * HD_ + half * 16];
    s0 = wmma_f16(aq0, *(const v16h*)kp0, s0);
    s0 = wmma_f16(aq1, *(const v16h*)(kp0 + 32), s0);
    s1 = wmma_f16(aq0, *(const v16h*)kp1, s1);
    s1 = wmma_f16(aq1, *(const v16h*)(kp1 + 32), s1);

    // ---- mask + scale + online softmax stats (row = 16-lane group)
    const int key0 = kb + col, key1 = kb + 16 + col;
    float p0[8], p1[8], rmax[8];
#pragma unroll
    for (int r = 0; r < 8; ++r) {
      const int q = qBase + r + half * 8;
      p0[r] = (key0 <= q) ? s0[r] * scale : -3.0e38f;
      p1[r] = (key1 <= q) ? s1[r] * scale : -3.0e38f;
      rmax[r] = fmaxf(p0[r], p1[r]);
    }
    for (int off = 1; off < 16; off <<= 1)
#pragma unroll
      for (int r = 0; r < 8; ++r) rmax[r] = fmaxf(rmax[r], __shfl_xor(rmax[r], off, 32));

    float alpha[8], rs[8];
#pragma unroll
    for (int r = 0; r < 8; ++r) {
      const float mnew = fmaxf(m_run[r], rmax[r]);
      alpha[r] = __expf(m_run[r] - mnew);
      m_run[r] = mnew;
      p0[r] = __expf(p0[r] - mnew);
      p1[r] = __expf(p1[r] - mnew);
      rs[r] = p0[r] + p1[r];
    }
    for (int off = 1; off < 16; off <<= 1)
#pragma unroll
      for (int r = 0; r < 8; ++r) rs[r] += __shfl_xor(rs[r], off, 32);
#pragma unroll
    for (int r = 0; r < 8; ++r) l_run[r] = l_run[r] * alpha[r] + rs[r];
#pragma unroll
    for (int t = 0; t < 4; ++t)
#pragma unroll
      for (int r = 0; r < 8; ++r) o[t][r] *= alpha[r];

    // ---- restage P (C layout -> A layout) via per-wave LDS (wave-ordered DS)
    _Float16* ps = &pst[wid][0];
#pragma unroll
    for (int r = 0; r < 8; ++r) {
      const int row = r + half * 8;
      ps[row * 32 + col]      = (_Float16)p0[r];
      ps[row * 32 + 16 + col] = (_Float16)p1[r];
    }
    v16h ap;
    {
      const _Float16* pr = ps + col * 32;
      *(v8h*)&ap       = *(const v8h*)(pr + half * 16);
      *((v8h*)&ap + 1) = *(const v8h*)(pr + half * 16 + 8);
    }

    // ---- O(16x64) += P(16x32) @ V(32x64) from LDS (B-operand contiguous)
#pragma unroll
    for (int t = 0; t < 4; ++t) {
      const _Float16* vp = &Vlds[cur][(size_t)(t * 16 + col) * 32 + half * 16];
      o[t] = wmma_f16(ap, *(const v16h*)vp, o[t]);
    }
  }

  // ---- normalize + write attention output [B,S,D] f16
#pragma unroll
  for (int r = 0; r < 8; ++r) {
    const float inv = 1.0f / l_run[r];
    const int qrowi = qBase + r + half * 8;
    _Float16* op = Ao + ((size_t)b * S_ + qrowi) * D_ + h * HD_;
#pragma unroll
    for (int t = 0; t < 4; ++t) op[t * 16 + col] = (_Float16)(o[t][r] * inv);
  }
}

// ---------------------------------------------------------------------------
// Output projection: Ao[8192,768]f16 @ WprojT[768,768]f16 + bias -> f32 out
// ---------------------------------------------------------------------------
__global__ __launch_bounds__(256)
void proj_gemm(const _Float16* __restrict__ X, const _Float16* __restrict__ Wt,
               const float* __restrict__ bias, float* __restrict__ out) {
  const int lane = threadIdx.x & 31;
  const int wave = blockIdx.x * (blockDim.x >> 5) + (threadIdx.x >> 5);
  const int nT = D_ / 64;
  const int mBase = (wave / nT) * 16;
  const int nBase = (wave % nT) * 64;
  const int col = lane & 15, half = lane >> 4;

  v8f acc[4];
#pragma unroll
  for (int t = 0; t < 4; ++t)
#pragma unroll
    for (int r = 0; r < 8; ++r) acc[t][r] = 0.0f;

  const _Float16* arow  = X + (size_t)(mBase + col) * D_;
  const _Float16* bbase = Wt + (size_t)(nBase + col) * D_ + half * 16;
  gemm_16x64(arow, bbase, col, half, acc);

#pragma unroll
  for (int t = 0; t < 4; ++t) {
    const int n = nBase + t * 16 + col;
    const float bv = bias[n];
#pragma unroll
    for (int r = 0; r < 8; ++r) {
      const int m = mBase + r + half * 8;
      out[(size_t)m * D_ + n] = acc[t][r] + bv;
    }
  }
}

// ---------------------------------------------------------------------------
extern "C" void kernel_launch(void* const* d_in, const int* in_sizes, int n_in,
                              void* d_out, int out_size, void* d_ws, size_t ws_size,
                              hipStream_t stream) {
  const float* x      = (const float*)d_in[0];
  const float* w_qkv  = (const float*)d_in[1];
  const float* b_qkv  = (const float*)d_in[2];
  const float* w_proj = (const float*)d_in[3];
  const float* b_proj = (const float*)d_in[4];
  float* out = (float*)d_out;

  char* ws = (char*)d_ws;
  size_t off = 0;
  auto alloc16 = [&](size_t elems) -> _Float16* {
    _Float16* p = (_Float16*)(ws + off);
    off += (elems * 2 + 255) & ~(size_t)255;
    return p;
  };
  _Float16* x16    = alloc16((size_t)M_ * D_);
  _Float16* wqkvT  = alloc16((size_t)N3_ * D_);
  _Float16* wprojT = alloc16((size_t)D_ * D_);
  _Float16* Qb     = alloc16((size_t)M_ * D_);    // [B,H,S,64]
  _Float16* Kb     = alloc16((size_t)M_ * D_);    // [B,H,S,64]
  _Float16* Vt     = alloc16((size_t)M_ * D_);    // [B,H,64,S]
  _Float16* Ao     = alloc16((size_t)M_ * D_);    // [B,S,D]

  { int n = M_ * D_;  cvt_f32_f16  <<<(n + 255) / 256, 256, 0, stream>>>(x, x16, n); }
  { int n = D_ * N3_; transpose_cvt<<<(n + 255) / 256, 256, 0, stream>>>(w_qkv, wqkvT, D_, N3_); }
  { int n = D_ * D_;  transpose_cvt<<<(n + 255) / 256, 256, 0, stream>>>(w_proj, wprojT, D_, D_); }

  { int waves = (M_ / 16) * (N3_ / 64);
    qkv_gemm<<<waves / 8, 256, 0, stream>>>(x16, wqkvT, b_qkv, Qb, Kb, Vt); }

  attn<<<B_ * H_ * (S_ / 64), 128, 0, stream>>>(Qb, Kb, Vt, Ao);

  { int waves = (M_ / 16) * (D_ / 64);
    proj_gemm<<<waves / 8, 256, 0, stream>>>(Ao, wprojT, b_proj, out); }
}
